// MultiHeadAttention_48687749267789
// MI455X (gfx1250) — matrix operand order repaired and checked by reading, not compiled, not run on hardware
//
#include <hip/hip_runtime.h>

typedef _Float16 f16;
typedef __attribute__((ext_vector_type(16))) _Float16 v16h;
typedef __attribute__((ext_vector_type(8)))  _Float16 v8h;
typedef __attribute__((ext_vector_type(8)))  float    v8f;
typedef int v4i __attribute__((vector_size(16)));

#define B_SZ  4
#define T_SZ  2048
#define C_SZ  1024
#define NH    16
#define HD    64
#define QK_SCALE 0.125f   // 1/sqrt(64)

#define AS1 __attribute__((address_space(1)))
#define AS3 __attribute__((address_space(3)))

#if __has_builtin(__builtin_amdgcn_global_load_async_to_lds_b128)
#define ASYNC_LDS 1
#else
#define ASYNC_LDS 0
#endif

// ---- 16-byte global -> LDS copy (per thread) ----
static __device__ inline void copy_b16B(f16* dst_lds, const f16* src_g) {
#if ASYNC_LDS
  __builtin_amdgcn_global_load_async_to_lds_b128(
      (AS1 v4i*)src_g, (AS3 v4i*)dst_lds, 0, 0);
#else
  *(v8h*)dst_lds = *(const v8h*)src_g;
#endif
}

static __device__ inline void wait_async_le1() {
  if (ASYNC_LDS) asm volatile("s_wait_asynccnt 0x1" ::: "memory");
}
static __device__ inline void wait_async_le0() {
  if (ASYNC_LDS) asm volatile("s_wait_asynccnt 0x0" ::: "memory");
}

// ---- A-fragment loader (16-bit A 16x32, contiguous k order) ----
// lanes 0-15 : row m=lane,    K {0..15}
// lanes 16-31: row m=lane-16, K {16..31}
static __device__ inline v16h load_a_frag(const f16* row, int kbase, int hi) {
  union { v16h v; v8h h[2]; } u;
  u.h[0] = *(const v8h*)(row + kbase + hi * 16);
  u.h[1] = *(const v8h*)(row + kbase + hi * 16 + 8);
  return u.v;
}

static __device__ inline v8f wmma_f16(v16h a, v16h b, v8f c) {
  return __builtin_amdgcn_wmma_f32_16x16x32_f16(false, a, false, b, (short)0, c,
                                                false, false);
}

// ---- elementwise f32 -> f16 conversion ----
__global__ void cvt_f32_f16(const float* __restrict__ s, f16* __restrict__ d, int n) {
  int i = blockIdx.x * blockDim.x + threadIdx.x;
  if (i < n) d[i] = (f16)s[i];
}

// =====================================================================
// Shared GEMM pipeline: block = 128M x 64N, 8 waves (each 16M x 64N).
// Two LDS B tiles (64N x 32K = 4KB each) with fully-peeled, branch-free
// double buffering: prologue issues copies for k-blocks 0/32, steady
// state runs 64-K double steps, tail consumes the last two blocks.
// =====================================================================
#define GEMM_COMPUTE(BufPtr, AFrag, Acc)                                        \
  {                                                                             \
    const f16* bb_ = (BufPtr);                                                  \
    v16h bfr[4];                                                                \
    _Pragma("unroll")                                                           \
    for (int nt = 0; nt < 4; ++nt)                                              \
      bfr[nt] = *(const v16h*)(bb_ + (nt * 16 + lm) * 32 + hi * 16);            \
    _Pragma("unroll")                                                           \
    for (int nt = 0; nt < 4; ++nt)                                              \
      Acc[nt] = wmma_f16((AFrag), bfr[nt], Acc[nt]);                            \
  }

#define GEMM_MAINLOOP(ARow, WPtr, NBase, Acc)                                   \
  {                                                                             \
    const int cn = tid >> 2;            /* 0..63 : local n row   */             \
    const int ck = (tid & 3) * 8;       /* 0/8/16/24 : k quarter */             \
    const f16* csrc = (WPtr) + (size_t)((NBase) + cn) * C_SZ + ck;              \
    f16* cd0 = &Bt[0][cn * 32 + ck];                                            \
    f16* cd1 = &Bt[1][cn * 32 + ck];                                            \
    /* prologue: k-blocks 0 and 32 in flight */                                 \
    copy_b16B(cd0, csrc);                                                       \
    copy_b16B(cd1, csrc + 32);                                                  \
    v16h aA0 = load_a_frag((ARow), 0, hi);                                      \
    v16h aA1 = load_a_frag((ARow), 32, hi);                                     \
    for (int kb = 0; kb < C_SZ - 64; kb += 64) {                                \
      wait_async_le1();                                                         \
      __syncthreads();                                                          \
      GEMM_COMPUTE(&Bt[0][0], aA0, Acc)                                         \
      __syncthreads();                                                          \
      copy_b16B(cd0, csrc + kb + 64);                                           \
      aA0 = load_a_frag((ARow), kb + 64, hi);                                   \
      __builtin_prefetch((ARow) + kb + 128, 0, 1);                              \
      wait_async_le1();                                                         \
      __syncthreads();                                                          \
      GEMM_COMPUTE(&Bt[1][0], aA1, Acc)                                         \
      __syncthreads();                                                          \
      copy_b16B(cd1, csrc + kb + 96);                                           \
      aA1 = load_a_frag((ARow), kb + 96, hi);                                   \
    }                                                                           \
    /* tail: last two k-blocks */                                               \
    wait_async_le1();                                                           \
    __syncthreads();                                                            \
    GEMM_COMPUTE(&Bt[0][0], aA0, Acc)                                           \
    wait_async_le0();                                                           \
    __syncthreads();                                                            \
    GEMM_COMPUTE(&Bt[1][0], aA1, Acc)                                           \
  }

// =====================================================================
// GEMM1: qkv = x @ w_qkv^T  (M=8192, N=3072, K=1024)
// epilogue scatters q (scaled) / k into [B,H,T,D], v into vT[B,H,D,T]
// =====================================================================
__global__ void __launch_bounds__(256)
gemm_qkv(const f16* __restrict__ xh, const f16* __restrict__ wh,
         f16* __restrict__ q, f16* __restrict__ k, f16* __restrict__ vT) {
  __shared__ __align__(32) f16 Bt[2][64 * 32];
  const int tid  = threadIdx.x;
  const int lane = tid & 31, wave = tid >> 5;
  const int hi = lane >> 4, lm = lane & 15;
  const int mbase = blockIdx.x * 128 + wave * 16;
  const int nbase = blockIdx.y * 64;

  const f16* arow = xh + (size_t)(mbase + lm) * C_SZ;
  v8f acc[4] = {};
  GEMM_MAINLOOP(arow, wh, nbase, acc)

#pragma unroll
  for (int nt = 0; nt < 4; ++nt) {
#pragma unroll
    for (int r = 0; r < 8; ++r) {
      int row = mbase + r + hi * 8;
      int bb  = row >> 11;
      int t   = row & (T_SZ - 1);
      int o   = nbase + nt * 16 + lm;
      int sec = o >> 10;
      int c   = o & (C_SZ - 1);
      int h   = c >> 6;
      int d   = c & 63;
      float v = acc[nt][r];
      if (sec == 0)
        q[((size_t)(bb * NH + h) * T_SZ + t) * HD + d] = (f16)(v * QK_SCALE);
      else if (sec == 1)
        k[((size_t)(bb * NH + h) * T_SZ + t) * HD + d] = (f16)v;
      else
        vT[((size_t)(bb * NH + h) * HD + d) * T_SZ + t] = (f16)v;
    }
  }
}

// =====================================================================
// Flash attention: 4 independent waves per block, each owns one 16-row
// q tile. KV streamed in 32-row chunks; online softmax; P re-laid-out
// through a private LDS slice (per-wave LDS ops are in-order: no barrier
// needed, only a compiler reorder fence).
// =====================================================================
__global__ void __launch_bounds__(128)
attn_fwd(const f16* __restrict__ q, const f16* __restrict__ k,
         const f16* __restrict__ vT, f16* __restrict__ o) {
  __shared__ __align__(32) f16 Plds[4][16 * 32];

  const int tid  = threadIdx.x;
  const int lane = tid & 31, wave = tid >> 5;
  const int hi = lane >> 4, lm = lane & 15;
  const int qt    = blockIdx.x * 4 + wave;
  const int bh    = blockIdx.y;          // b*NH + h
  const int qbase = qt * 16;
  f16* pl = &Plds[wave][0];

  const f16* qhd = q  + (size_t)bh * T_SZ * HD;
  const f16* khd = k  + (size_t)bh * T_SZ * HD;
  const f16* vhd = vT + (size_t)bh * HD * T_SZ;

  const f16* qrow = qhd + (size_t)(qbase + lm) * HD;
  v16h aQ0 = load_a_frag(qrow, 0, hi);
  v16h aQ1 = load_a_frag(qrow, 32, hi);

  v8f accO[4] = {};
  float rm[8], rl[8];
#pragma unroll
  for (int r = 0; r < 8; ++r) { rm[r] = -3.0e38f; rl[r] = 0.f; }

  for (int j = 0; j < qbase + 16; j += 32) {
    // ---- issue all K and V fragment loads up-front (clause + overlap) ----
    const f16* k0 = khd + (size_t)(j + lm) * HD + hi * 16;
    const f16* k1 = khd + (size_t)(j + 16 + lm) * HD + hi * 16;
    v16h bK00 = *(const v16h*)k0;
    v16h bK01 = *(const v16h*)(k0 + 32);
    v16h bK10 = *(const v16h*)k1;
    v16h bK11 = *(const v16h*)(k1 + 32);
    v16h bV[4];
#pragma unroll
    for (int c4 = 0; c4 < 4; ++c4)
      bV[c4] = *(const v16h*)(vhd + (size_t)(c4 * 16 + lm) * T_SZ + j + hi * 16);

    // ---- S = Q K^T (16 x 32) ----
    v8f s0 = {}, s1 = {};
    s0 = wmma_f16(aQ0, bK00, s0);
    s0 = wmma_f16(aQ1, bK01, s0);
    s1 = wmma_f16(aQ0, bK10, s1);
    s1 = wmma_f16(aQ1, bK11, s1);

    // ---- causal mask + online softmax (row r at m = r + 8*hi) ----
    float scl[8];
#pragma unroll
    for (int r = 0; r < 8; ++r) {
      int qi = qbase + r + hi * 8;
      float a0 = (j + lm      > qi) ? -3.0e38f : s0[r];
      float a1 = (j + 16 + lm > qi) ? -3.0e38f : s1[r];
      float mx = fmaxf(a0, a1);
#pragma unroll
      for (int d = 1; d < 16; d <<= 1) mx = fmaxf(mx, __shfl_xor(mx, d, 32));
      mx = fmaxf(mx, rm[r]);
      scl[r] = __expf(rm[r] - mx);
      rm[r] = mx;
      float p0 = __expf(a0 - mx);
      float p1 = __expf(a1 - mx);
      float ps = p0 + p1;
#pragma unroll
      for (int d = 1; d < 16; d <<= 1) ps += __shfl_xor(ps, d, 32);
      rl[r] = rl[r] * scl[r] + ps;
      pl[(r + hi * 8) * 32 + lm]      = (f16)p0;
      pl[(r + hi * 8) * 32 + 16 + lm] = (f16)p1;
#pragma unroll
      for (int c4 = 0; c4 < 4; ++c4) accO[c4][r] *= scl[r];
    }
    asm volatile("" ::: "memory");   // keep ds stores before ds loads (HW is in-order per wave)

    // ---- O += P V ----
    v16h aP = load_a_frag(pl + lm * 32, 0, hi);
#pragma unroll
    for (int c4 = 0; c4 < 4; ++c4)
      accO[c4] = wmma_f16(aP, bV[c4], accO[c4]);
    asm volatile("" ::: "memory");
  }

  // ---- normalize + store f16 into attn[B,T,C] (row-contiguous) ----
  const int h  = bh & (NH - 1);
  const int bb = bh >> 4;
#pragma unroll
  for (int r = 0; r < 8; ++r) {
    float inv = 1.0f / rl[r];
    int t = qbase + r + hi * 8;
    f16* orow = o + ((size_t)bb * T_SZ + t) * C_SZ + h * HD;
#pragma unroll
    for (int c4 = 0; c4 < 4; ++c4)
      orow[c4 * 16 + lm] = (f16)(accO[c4][r] * inv);
  }
}

// =====================================================================
// GEMM2: out = attn @ w_out^T + b_out  (M=8192, N=1024, K=1024), f32 out
// =====================================================================
__global__ void __launch_bounds__(256)
gemm_out(const f16* __restrict__ ah, const f16* __restrict__ wh,
         const float* __restrict__ bias, float* __restrict__ out) {
  __shared__ __align__(32) f16 Bt[2][64 * 32];
  const int tid  = threadIdx.x;
  const int lane = tid & 31, wave = tid >> 5;
  const int hi = lane >> 4, lm = lane & 15;
  const int mbase = blockIdx.x * 128 + wave * 16;
  const int nbase = blockIdx.y * 64;

  const f16* arow = ah + (size_t)(mbase + lm) * C_SZ;
  v8f acc[4] = {};
  GEMM_MAINLOOP(arow, wh, nbase, acc)

#pragma unroll
  for (int nt = 0; nt < 4; ++nt) {
#pragma unroll
    for (int r = 0; r < 8; ++r) {
      int row = mbase + r + hi * 8;
      int col = nbase + nt * 16 + lm;
      out[(size_t)row * C_SZ + col] = acc[nt][r] + bias[col];
    }
  }
}

// =====================================================================
extern "C" void kernel_launch(void* const* d_in, const int* in_sizes, int n_in,
                              void* d_out, int out_size, void* d_ws, size_t ws_size,
                              hipStream_t stream) {
  const float* x     = (const float*)d_in[0];
  const float* w_qkv = (const float*)d_in[1];
  const float* w_out = (const float*)d_in[2];
  const float* b_out = (const float*)d_in[3];
  float* out = (float*)d_out;

  char*  ws  = (char*)d_ws;
  size_t off = 0;
  const size_t EL = (size_t)B_SZ * T_SZ * C_SZ;   // 8M elements

  f16* xh    = (f16*)(ws + off); off += EL * 2;
  f16* wqkvh = (f16*)(ws + off); off += (size_t)3 * C_SZ * C_SZ * 2;
  f16* wouth = (f16*)(ws + off); off += (size_t)C_SZ * C_SZ * 2;
  f16* qb    = (f16*)(ws + off); off += EL * 2;
  f16* kbuf  = (f16*)(ws + off); off += EL * 2;
  f16* vTb   = (f16*)(ws + off); off += EL * 2;
  f16* attnb = (f16*)(ws + off); off += EL * 2;
  (void)ws_size; (void)in_sizes; (void)n_in; (void)out_size;

  int nx = (int)EL;
  cvt_f32_f16<<<(nx + 255) / 256, 256, 0, stream>>>(x, xh, nx);
  int nw = 3 * C_SZ * C_SZ;
  cvt_f32_f16<<<(nw + 255) / 256, 256, 0, stream>>>(w_qkv, wqkvh, nw);
  int no = C_SZ * C_SZ;
  cvt_f32_f16<<<(no + 255) / 256, 256, 0, stream>>>(w_out, wouth, no);

  gemm_qkv<<<dim3(B_SZ * T_SZ / 128, (3 * C_SZ) / 64), 256, 0, stream>>>(
      xh, wqkvh, qb, kbuf, vTb);

  attn_fwd<<<dim3(T_SZ / 64, B_SZ * NH), 128, 0, stream>>>(qb, kbuf, vTb, attnb);

  gemm_out<<<dim3(B_SZ * T_SZ / 128, C_SZ / 64), 256, 0, stream>>>(
      attnb, wouth, b_out, out);
}
